// Causal_Attention_2929167695943
// MI455X (gfx1250) — matrix operand order repaired and checked by reading, not compiled, not run on hardware
//
#include <hip/hip_runtime.h>

#define DIM   1024
#define SEQ   2048
#define BATCH 4

typedef __attribute__((ext_vector_type(16))) __bf16 v16bf;
typedef __attribute__((ext_vector_type(8)))  __bf16 v8bf;
typedef __attribute__((ext_vector_type(4)))  __bf16 v4bf;
typedef __attribute__((ext_vector_type(8)))  float  v8f;
typedef __attribute__((ext_vector_type(4)))  float  v4f;

__device__ inline __bf16 f2bf(float f) { return (__bf16)f; }   // v_cvt_pk_bf16_f32
__device__ inline float bf2f(__bf16 b) {
  unsigned short s = __builtin_bit_cast(unsigned short, b);
  unsigned u = ((unsigned)s) << 16;
  return __builtin_bit_cast(float, u);
}

// Shared GEMM geometry: M_TILE=256, N_TILE=64, K_STEP=32.
// 8 waves; each wave owns 32 rows x 64 cols = 2 row-groups x 4 col-tiles.
// Per K-step per wave: 4 A ds_load_b128 + 8 B ds_load_b128 -> 8 WMMAs
// issued back-to-back (B fragments reused across both row groups).

// ---------------------------------------------------------------------------
// Kernel 1: QKV projection.  Y[8192,1024](bf16) = X[8192,1024](f32) @ W (f32)
// ---------------------------------------------------------------------------
__global__ __launch_bounds__(256) void qkv_proj_kernel(
    const float* __restrict__ X,
    const float* __restrict__ Wq,
    const float* __restrict__ Wk,
    const float* __restrict__ Wv,
    __bf16* __restrict__ Q,
    __bf16* __restrict__ K,
    __bf16* __restrict__ V)
{
  __shared__ __bf16 Asl[2][256 * 32];   // [m][k]
  __shared__ __bf16 Bsl[2][64 * 32];    // [n][k] (transposed W tile)

  const float* W; __bf16* Y; float scale;
  if (blockIdx.z == 0)      { W = Wq; Y = Q; scale = 0.03125f; }  // 1/sqrt(1024)
  else if (blockIdx.z == 1) { W = Wk; Y = K; scale = 1.0f; }
  else                      { W = Wv; Y = V; scale = 1.0f; }

  const int m0   = blockIdx.x * 256;
  const int n0   = blockIdx.y * 64;
  const int tid  = threadIdx.x;
  const int lane = tid & 31;
  const int wave = tid >> 5;
  const int fr   = lane & 15;
  const int hi   = lane >> 4;

  v8f acc[2][4];
#pragma unroll
  for (int rg = 0; rg < 2; rg++)
#pragma unroll
    for (int t = 0; t < 4; t++)
#pragma unroll
      for (int j = 0; j < 8; j++) acc[rg][t][j] = 0.0f;

  auto stage = [&](int p, int k0) {
    // X tile 256x32: f32 -> bf16, 32 elems/thread as 2 units of 16
#pragma unroll
    for (int u = 0; u < 2; u++) {
      int rr = u * 128 + (tid >> 1);
      int hh = (tid & 1) * 16;
      const float* xp = X + (size_t)(m0 + rr) * DIM + k0 + hh;
#pragma unroll
      for (int i = 0; i < 4; i++) {
        v4f x4 = *(const v4f*)(xp + i * 4);
        v4bf b4;
#pragma unroll
        for (int j = 0; j < 4; j++) b4[j] = f2bf(x4[j]);
        *(v4bf*)&Asl[p][rr * 32 + hh + i * 4] = b4;
      }
    }
    // W tile 32x64 -> transposed LDS [n][k]
#pragma unroll
    for (int i = 0; i < 2; i++) {
      int e  = i * 256 + tid;            // 0..511
      int kk = e >> 4;                   // 0..31
      int nn = (e & 15) * 4;             // 0..60
      v4f w4 = *(const v4f*)(W + (size_t)(k0 + kk) * DIM + n0 + nn);
#pragma unroll
      for (int j = 0; j < 4; j++) Bsl[p][(nn + j) * 32 + kk] = f2bf(w4[j]);
    }
    if (k0 + 32 < DIM) {
      __builtin_prefetch(X + (size_t)(m0 + (tid >> 1)) * DIM + k0 + 32, 0, 0);
      __builtin_prefetch(W + (size_t)(k0 + 32 + (tid >> 4)) * DIM + n0, 0, 0);
    }
  };

  auto compute = [&](int p) {
    // Issue all LDS loads up front, then stream 8 WMMAs.
    const __bf16* a0p = &Asl[p][(wave * 32 + fr) * 32 + hi * 16];
    const __bf16* a1p = a0p + 16 * 32;
    v8bf a0lo = *(const v8bf*)a0p,        a0hi = *(const v8bf*)(a0p + 8);
    v8bf a1lo = *(const v8bf*)a1p,        a1hi = *(const v8bf*)(a1p + 8);
    v16bf bfr[4];
#pragma unroll
    for (int t = 0; t < 4; t++)
      bfr[t] = *(const v16bf*)&Bsl[p][(t * 16 + fr) * 32 + hi * 16];
    v16bf af0, af1;
#pragma unroll
    for (int j = 0; j < 8; j++) {
      af0[j] = a0lo[j]; af0[j + 8] = a0hi[j];
      af1[j] = a1lo[j]; af1[j + 8] = a1hi[j];
    }
#pragma unroll
    for (int t = 0; t < 4; t++)
      acc[0][t] = __builtin_amdgcn_wmma_f32_16x16x32_bf16(
          false, af0, false, bfr[t], (short)0, acc[0][t], false, false);
#pragma unroll
    for (int t = 0; t < 4; t++)
      acc[1][t] = __builtin_amdgcn_wmma_f32_16x16x32_bf16(
          false, af1, false, bfr[t], (short)0, acc[1][t], false, false);
  };

  stage(0, 0);
  int p = 0;
  for (int k0 = 0; k0 < DIM; k0 += 32) {
    __syncthreads();
    if (k0 + 32 < DIM) stage(p ^ 1, k0 + 32);
    compute(p);
    p ^= 1;
  }

#pragma unroll
  for (int rg = 0; rg < 2; rg++) {
    const int orow = m0 + wave * 32 + rg * 16 + hi * 8;
#pragma unroll
    for (int t = 0; t < 4; t++)
#pragma unroll
      for (int g = 0; g < 8; g++)
        Y[(size_t)(orow + g) * DIM + n0 + fr + t * 16] = f2bf(acc[rg][t][g] * scale);
  }
}

// ---------------------------------------------------------------------------
// Kernel 2: scores S[b,q,k] = Q[b,q,:] . K[b,k,:]  (Q pre-scaled), causal mask.
// ---------------------------------------------------------------------------
__global__ __launch_bounds__(256) void scores_kernel(
    const __bf16* __restrict__ Q,
    const __bf16* __restrict__ K,
    __bf16* __restrict__ Sc)
{
  __shared__ __bf16 Asl[2][256 * 32];   // Q tile [q][d]
  __shared__ __bf16 Bsl[2][64 * 32];    // K tile [k][d]

  const int m0 = blockIdx.x * 256;   // query rows
  const int n0 = blockIdx.y * 64;    // key cols
  const int b  = blockIdx.z;
  const int tid = threadIdx.x, lane = tid & 31, wave = tid >> 5;
  const int fr = lane & 15, hi = lane >> 4;

  __bf16* S = Sc + (size_t)b * SEQ * SEQ;

  if (n0 >= m0 + 256) {              // entire tile above the diagonal
    v8bf mv;
#pragma unroll
    for (int j = 0; j < 8; j++) mv[j] = f2bf(-1.0e30f);
#pragma unroll
    for (int i = 0; i < 8; i++) {    // 256x64 = 2048 v8bf units
      int e = i * 256 + tid;
      int rr = e >> 3, cc = (e & 7) * 8;
      *(v8bf*)&S[(size_t)(m0 + rr) * SEQ + n0 + cc] = mv;
    }
    return;
  }

  const __bf16* Qb = Q + (size_t)b * SEQ * DIM;
  const __bf16* Kb = K + (size_t)b * SEQ * DIM;

  v8f acc[2][4];
#pragma unroll
  for (int rg = 0; rg < 2; rg++)
#pragma unroll
    for (int t = 0; t < 4; t++)
#pragma unroll
      for (int j = 0; j < 8; j++) acc[rg][t][j] = 0.0f;

  const int nB = tid >> 2;           // B row (key)
  const int pB = (tid & 3) * 8;

  auto stage = [&](int p, int k0) {
#pragma unroll
    for (int u = 0; u < 2; u++) {
      int rr = u * 128 + (tid >> 1);
      int hh = (tid & 1) * 16;
      const v8bf* qp = (const v8bf*)(Qb + (size_t)(m0 + rr) * DIM + k0 + hh);
      *(v8bf*)&Asl[p][rr * 32 + hh]     = qp[0];
      *(v8bf*)&Asl[p][rr * 32 + hh + 8] = qp[1];
    }
    *(v8bf*)&Bsl[p][nB * 32 + pB] =
        *(const v8bf*)(Kb + (size_t)(n0 + nB) * DIM + k0 + pB);
    if (k0 + 32 < DIM) {
      __builtin_prefetch(Qb + (size_t)(m0 + (tid >> 1)) * DIM + k0 + 32, 0, 0);
      __builtin_prefetch(Kb + (size_t)(n0 + nB) * DIM + k0 + 32 + pB, 0, 0);
    }
  };

  auto compute = [&](int p) {
    const __bf16* a0p = &Asl[p][(wave * 32 + fr) * 32 + hi * 16];
    const __bf16* a1p = a0p + 16 * 32;
    v8bf a0lo = *(const v8bf*)a0p,        a0hi = *(const v8bf*)(a0p + 8);
    v8bf a1lo = *(const v8bf*)a1p,        a1hi = *(const v8bf*)(a1p + 8);
    v16bf bfr[4];
#pragma unroll
    for (int t = 0; t < 4; t++)
      bfr[t] = *(const v16bf*)&Bsl[p][(t * 16 + fr) * 32 + hi * 16];
    v16bf af0, af1;
#pragma unroll
    for (int j = 0; j < 8; j++) {
      af0[j] = a0lo[j]; af0[j + 8] = a0hi[j];
      af1[j] = a1lo[j]; af1[j + 8] = a1hi[j];
    }
#pragma unroll
    for (int t = 0; t < 4; t++)
      acc[0][t] = __builtin_amdgcn_wmma_f32_16x16x32_bf16(
          false, af0, false, bfr[t], (short)0, acc[0][t], false, false);
#pragma unroll
    for (int t = 0; t < 4; t++)
      acc[1][t] = __builtin_amdgcn_wmma_f32_16x16x32_bf16(
          false, af1, false, bfr[t], (short)0, acc[1][t], false, false);
  };

  stage(0, 0);
  int p = 0;
  for (int k0 = 0; k0 < DIM; k0 += 32) {
    __syncthreads();
    if (k0 + 32 < DIM) stage(p ^ 1, k0 + 32);
    compute(p);
    p ^= 1;
  }

#pragma unroll
  for (int rg = 0; rg < 2; rg++) {
    const int q0r = m0 + wave * 32 + rg * 16 + hi * 8;
#pragma unroll
    for (int t = 0; t < 4; t++)
#pragma unroll
      for (int g = 0; g < 8; g++) {
        int qq = q0r + g;
        int kk = n0 + t * 16 + fr;
        float sv = (kk <= qq) ? acc[rg][t][g] : -1.0e30f;
        S[(size_t)qq * SEQ + kk] = f2bf(sv);
      }
  }
}

// ---------------------------------------------------------------------------
// Kernel 3: per-row softmax stats: m = max_k s, zinv = 1/sum exp(s-m).
// ---------------------------------------------------------------------------
__global__ __launch_bounds__(256) void row_stats_kernel(
    const __bf16* __restrict__ Sc,
    float* __restrict__ stats)
{
  const int row  = blockIdx.x * 8 + (threadIdx.x >> 5);   // 0..8191
  const int lane = threadIdx.x & 31;
  const __bf16* s = Sc + (size_t)row * SEQ;

  float m = -3.0e38f;
  for (int i = 0; i < SEQ; i += 256) {
    v8bf v = *(const v8bf*)(s + i + lane * 8);
#pragma unroll
    for (int j = 0; j < 8; j++) m = fmaxf(m, bf2f(v[j]));
  }
#pragma unroll
  for (int off = 16; off > 0; off >>= 1) m = fmaxf(m, __shfl_xor(m, off, 32));

  float z = 0.0f;
  for (int i = 0; i < SEQ; i += 256) {
    v8bf v = *(const v8bf*)(s + i + lane * 8);
#pragma unroll
    for (int j = 0; j < 8; j++) z += __expf(bf2f(v[j]) - m);
  }
#pragma unroll
  for (int off = 16; off > 0; off >>= 1) z += __shfl_xor(z, off, 32);

  if (lane == 0) {
    stats[(size_t)row * 2 + 0] = m;
    stats[(size_t)row * 2 + 1] = 1.0f / z;
  }
}

// ---------------------------------------------------------------------------
// Kernel 4: context = softmax(S) @ V; P computed on the fly while staging.
// Key loop bounded by m0+256 (causal).
// ---------------------------------------------------------------------------
__global__ __launch_bounds__(256) void context_kernel(
    const __bf16* __restrict__ Sc,
    const float* __restrict__ stats,
    const __bf16* __restrict__ V,
    float* __restrict__ Out)
{
  __shared__ __bf16 Asl[2][256 * 32];   // P tile [q][k]
  __shared__ __bf16 Bsl[2][64 * 32];    // V tile transposed [n][k]

  const int m0 = blockIdx.x * 256;   // query rows
  const int n0 = blockIdx.y * 64;    // output dims
  const int b  = blockIdx.z;
  const int tid = threadIdx.x, lane = tid & 31, wave = tid >> 5;
  const int fr = lane & 15, hi = lane >> 4;

  const __bf16* S  = Sc + (size_t)b * SEQ * SEQ;
  const __bf16* Vb = V  + (size_t)b * SEQ * DIM;

  // Each thread stages 2 score rows: tid>>1 and 128+(tid>>1)
  float mrow[2], zrow[2];
#pragma unroll
  for (int u = 0; u < 2; u++) {
    int rr = u * 128 + (tid >> 1);
    mrow[u] = stats[((size_t)b * SEQ + m0 + rr) * 2 + 0];
    zrow[u] = stats[((size_t)b * SEQ + m0 + rr) * 2 + 1];
  }

  const int kV = tid >> 3;           // 0..31
  const int nV = (tid & 7) * 8;      // 0..56

  v8f acc[2][4];
#pragma unroll
  for (int rg = 0; rg < 2; rg++)
#pragma unroll
    for (int t = 0; t < 4; t++)
#pragma unroll
      for (int j = 0; j < 8; j++) acc[rg][t][j] = 0.0f;

  const int kmax = m0 + 256;         // causal: keys <= max query in tile

  auto stage = [&](int p, int k0) {
#pragma unroll
    for (int u = 0; u < 2; u++) {
      int rr = u * 128 + (tid >> 1);
      int hh = (tid & 1) * 16;
      const v8bf* sp = (const v8bf*)(S + (size_t)(m0 + rr) * SEQ + k0 + hh);
      v8bf s0 = sp[0], s1 = sp[1];
      v8bf p0, p1;
#pragma unroll
      for (int j = 0; j < 8; j++) {
        p0[j] = f2bf(__expf(bf2f(s0[j]) - mrow[u]) * zrow[u]);
        p1[j] = f2bf(__expf(bf2f(s1[j]) - mrow[u]) * zrow[u]);
      }
      *(v8bf*)&Asl[p][rr * 32 + hh]     = p0;
      *(v8bf*)&Asl[p][rr * 32 + hh + 8] = p1;
    }
    v8bf v8 = *(const v8bf*)(Vb + (size_t)(k0 + kV) * DIM + n0 + nV);
#pragma unroll
    for (int j = 0; j < 8; j++) Bsl[p][(nV + j) * 32 + kV] = v8[j];
    if (k0 + 32 < kmax) {
      __builtin_prefetch(S + (size_t)(m0 + (tid >> 1)) * SEQ + k0 + 32, 0, 0);
      __builtin_prefetch(Vb + (size_t)(k0 + 32 + kV) * DIM + n0 + nV, 0, 0);
    }
  };

  auto compute = [&](int p) {
    const __bf16* a0p = &Asl[p][(wave * 32 + fr) * 32 + hi * 16];
    const __bf16* a1p = a0p + 16 * 32;
    v8bf a0lo = *(const v8bf*)a0p,        a0hi = *(const v8bf*)(a0p + 8);
    v8bf a1lo = *(const v8bf*)a1p,        a1hi = *(const v8bf*)(a1p + 8);
    v16bf bfr[4];
#pragma unroll
    for (int t = 0; t < 4; t++)
      bfr[t] = *(const v16bf*)&Bsl[p][(t * 16 + fr) * 32 + hi * 16];
    v16bf af0, af1;
#pragma unroll
    for (int j = 0; j < 8; j++) {
      af0[j] = a0lo[j]; af0[j + 8] = a0hi[j];
      af1[j] = a1lo[j]; af1[j + 8] = a1hi[j];
    }
#pragma unroll
    for (int t = 0; t < 4; t++)
      acc[0][t] = __builtin_amdgcn_wmma_f32_16x16x32_bf16(
          false, af0, false, bfr[t], (short)0, acc[0][t], false, false);
#pragma unroll
    for (int t = 0; t < 4; t++)
      acc[1][t] = __builtin_amdgcn_wmma_f32_16x16x32_bf16(
          false, af1, false, bfr[t], (short)0, acc[1][t], false, false);
  };

  stage(0, 0);
  int p = 0;
  for (int k0 = 0; k0 < kmax; k0 += 32) {
    __syncthreads();
    if (k0 + 32 < kmax) stage(p ^ 1, k0 + 32);
    compute(p);
    p ^= 1;
  }

#pragma unroll
  for (int rg = 0; rg < 2; rg++) {
    const int orow = m0 + wave * 32 + rg * 16 + hi * 8;
#pragma unroll
    for (int t = 0; t < 4; t++)
#pragma unroll
      for (int g = 0; g < 8; g++)
        Out[((size_t)b * SEQ + orow + g) * DIM + n0 + fr + t * 16] = acc[rg][t][g];
  }
}

// ---------------------------------------------------------------------------
// Host launch.  Workspace layout (bytes):
//   [0,16M)   Q bf16      [16M,32M) K bf16     [32M,48M) V bf16
//   [48M,80M) scores bf16 [80M, +64K) row stats f32 (m, 1/z)
// ---------------------------------------------------------------------------
extern "C" void kernel_launch(void* const* d_in, const int* in_sizes, int n_in,
                              void* d_out, int out_size, void* d_ws, size_t ws_size,
                              hipStream_t stream) {
  (void)in_sizes; (void)n_in; (void)out_size; (void)ws_size;

  const float* x  = (const float*)d_in[0];
  const float* Wq = (const float*)d_in[1];
  const float* Wk = (const float*)d_in[2];
  const float* Wv = (const float*)d_in[3];
  float* out = (float*)d_out;

  char* ws = (char*)d_ws;
  __bf16* Qb = (__bf16*)(ws);
  __bf16* Kb = (__bf16*)(ws + (size_t)16 * 1024 * 1024);
  __bf16* Vb = (__bf16*)(ws + (size_t)32 * 1024 * 1024);
  __bf16* Sc = (__bf16*)(ws + (size_t)48 * 1024 * 1024);
  float*  st = (float*) (ws + (size_t)80 * 1024 * 1024);

  dim3 blk(256);
  qkv_proj_kernel<<<dim3(32, 16, 3), blk, 0, stream>>>(x, Wq, Wk, Wv, Qb, Kb, Vb);
  scores_kernel<<<dim3(8, 32, 4), blk, 0, stream>>>(Qb, Kb, Sc);
  row_stats_kernel<<<dim3(1024), blk, 0, stream>>>(Sc, st);
  context_kernel<<<dim3(8, 16, 4), blk, 0, stream>>>(Sc, st, Vb, out);
}
